// RelativeGlobalAttention_69552700391964
// MI455X (gfx1250) — matrix operand order repaired and checked by reading, not compiled, not run on hardware
//
#include <hip/hip_runtime.h>

// Problem constants (fixed by the reference)
#define BATCH   2
#define SEQ     1024
#define DMODEL  1024
#define NHEADS  16
#define DK      64
#define MROWS   (BATCH * SEQ)        // 2048 rows for all projection GEMMs
#define ER_ROWS 2047
#define ER_PAD  2048                 // pad with one zero row so the rel band never reads OOB

typedef __bf16 bf16;
typedef bf16  v16bf __attribute__((ext_vector_type(16)));
typedef bf16  v8bf  __attribute__((ext_vector_type(8)));
typedef float v8f   __attribute__((ext_vector_type(8)));

// ---------------------------------------------------------------------------
// WMMA fragment helpers (wave32 layouts per CDNA5 ISA 7.12.2)
// ---------------------------------------------------------------------------

// A-matrix 16x32 bf16: lanes 0-15 hold M=lane, K groups {0..7, 16..23};
// lanes 16-31 hold M=lane-16, K groups {8..15, 24..31}.
__device__ __forceinline__ v16bf load_a_frag(const bf16* __restrict__ base,
                                             int ld, int kk0, int lane) {
  const int m  = lane & 15;
  const int kh = (lane < 16) ? 0 : 8;
  const bf16* p = base + (size_t)m * ld + kk0 + kh;
  v8bf lo = *(const v8bf*)(p);        // K = kk0+kh .. +7
  v8bf hi = *(const v8bf*)(p + 16);   // K = kk0+kh+16 .. +23
  v16bf r;
#pragma unroll
  for (int i = 0; i < 8; ++i) { r[i] = lo[i]; r[i + 8] = hi[i]; }
  return r;
}

// B-matrix 32x16 bf16: lane n = lane&15; lanes 0-15 hold K groups {0..7, 16..23},
// lanes 16-31 hold K groups {8..15, 24..31}. Source element B(k,n) = base[n*ld + k]
// (i.e. source matrix rows indexed by n, contiguous in k) -> two 16B loads.
__device__ __forceinline__ v16bf load_b_frag(const bf16* __restrict__ base,
                                             int ld, int lane) {
  const int n  = lane & 15;
  const int kb = (lane < 16) ? 0 : 8;
  const bf16* p = base + (size_t)n * ld + kb;
  v8bf lo = *(const v8bf*)(p);
  v8bf hi = *(const v8bf*)(p + 16);
  v16bf r;
#pragma unroll
  for (int i = 0; i < 8; ++i) { r[i] = lo[i]; r[i + 8] = hi[i]; }
  return r;
}

__device__ __forceinline__ v8f wmma_bf16(v16bf a, v16bf b, v8f c) {
  return __builtin_amdgcn_wmma_f32_16x16x32_bf16(
      /*neg_a=*/false, a, /*neg_b=*/false, b,
      /*c_mod=*/(short)0, c, /*reuse_a=*/false, /*reuse_b=*/false);
}

// ---------------------------------------------------------------------------
// fp32 -> bf16 conversion (with zero padding past n_in, used for Er pad row)
// ---------------------------------------------------------------------------
__global__ void cvt_f32_bf16(const float* __restrict__ in, bf16* __restrict__ out,
                             int n_in, int n_out) {
  int i = blockIdx.x * blockDim.x + threadIdx.x;
  if (i < n_out) out[i] = (i < n_in) ? (bf16)in[i] : (bf16)0.0f;
}

// ---------------------------------------------------------------------------
// GEMM:  out[m,n] = sum_k A[m,k] * W[n,k]  + bias[n]      (Y = X @ W^T + b)
// One wave computes a 16(M) x 64(N) tile; K = DMODEL in steps of 32.
// mode 0: store bf16 row-major [MROWS, DMODEL]
// mode 1: store bf16 transposed per-head: Vt[((b*H+h)*DK+d)*SEQ + s]
// mode 2: store fp32 row-major (final output)
// ---------------------------------------------------------------------------
__global__ __launch_bounds__(32) void gemm16x64_bf16(
    const bf16* __restrict__ A, const bf16* __restrict__ W,
    const float* __restrict__ bias,
    bf16* __restrict__ outb, float* __restrict__ outf, int mode) {
  const int lane = threadIdx.x;
  const int tile = blockIdx.x;
  const int m0 = (tile & (MROWS / 16 - 1)) * 16;
  const int n0 = (tile / (MROWS / 16)) * 64;
  const bf16* Ab = A + (size_t)m0 * DMODEL;

  v8f acc0 = {}, acc1 = {}, acc2 = {}, acc3 = {};
  for (int k = 0; k < DMODEL; k += 32) {
    // Prefetch next K-step lines into the WGP cache (global_prefetch_b8).
    if (k + 32 < DMODEL) {
      const int kn = k + 32;
      __builtin_prefetch(Ab + (size_t)(lane & 15) * DMODEL + kn, 0, 1);
      __builtin_prefetch(W + (size_t)(n0 + (lane & 15)) * DMODEL + kn, 0, 1);
      __builtin_prefetch(W + (size_t)(n0 + 16 + (lane & 15)) * DMODEL + kn, 0, 1);
      __builtin_prefetch(W + (size_t)(n0 + 32 + (lane & 15)) * DMODEL + kn, 0, 1);
      __builtin_prefetch(W + (size_t)(n0 + 48 + (lane & 15)) * DMODEL + kn, 0, 1);
    }
    v16bf a  = load_a_frag(Ab, DMODEL, k, lane);
    v16bf b0 = load_b_frag(W + (size_t)(n0 +  0) * DMODEL + k, DMODEL, lane);
    v16bf b1 = load_b_frag(W + (size_t)(n0 + 16) * DMODEL + k, DMODEL, lane);
    v16bf b2 = load_b_frag(W + (size_t)(n0 + 32) * DMODEL + k, DMODEL, lane);
    v16bf b3 = load_b_frag(W + (size_t)(n0 + 48) * DMODEL + k, DMODEL, lane);
    acc0 = wmma_bf16(a, b0, acc0);
    acc1 = wmma_bf16(a, b1, acc1);
    acc2 = wmma_bf16(a, b2, acc2);
    acc3 = wmma_bf16(a, b3, acc3);
  }

  const int nlo = lane & 15;
  const int mhi = (lane & 16) ? 8 : 0;
  v8f accs[4] = {acc0, acc1, acc2, acc3};
#pragma unroll
  for (int nt = 0; nt < 4; ++nt) {
    const int   ncol = n0 + nt * 16 + nlo;
    const float badd = bias ? bias[ncol] : 0.0f;
#pragma unroll
    for (int r = 0; r < 8; ++r) {
      const int   mrow = m0 + r + mhi;
      const float val  = accs[nt][r] + badd;
      if (mode == 0) {
        outb[(size_t)mrow * DMODEL + ncol] = (bf16)val;
      } else if (mode == 1) {
        const int b = mrow >> 10, s = mrow & (SEQ - 1);
        const int h = ncol >> 6,  d = ncol & (DK - 1);
        outb[(((size_t)(b * NHEADS + h)) * DK + d) * SEQ + s] = (bf16)val;
      } else {
        outf[(size_t)mrow * DMODEL + ncol] = val;
      }
    }
  }
}

// ---------------------------------------------------------------------------
// Flash attention with banded relative-position term.
// One wave per (b, h, 16 query rows); streams K in 32-column tiles.
//   scores = (Q K^T + Qer_band) * 1/sqrt(64), online softmax, O = P V.
// Gbuf: 16x48 band GEMM result; qer[m,kk] = Gbuf[m][m-kk+31].
// ---------------------------------------------------------------------------
__global__ __launch_bounds__(32) void attn_rel_flash(
    const bf16* __restrict__ Q, const bf16* __restrict__ Kp,
    const bf16* __restrict__ Vt, const bf16* __restrict__ Er,
    bf16* __restrict__ Ctx) {
  __shared__ float Gbuf[16][48];
  __shared__ bf16  Pbuf[16][32];

  const int blk = blockIdx.x;
  const int qt = blk & 63;
  const int h  = (blk >> 6) & (NHEADS - 1);
  const int b  = blk >> 10;
  const int q0 = qt * 16;
  const int lane = threadIdx.x;
  const int nlo  = lane & 15;
  const int mhi  = (lane & 16) ? 8 : 0;

  const bf16* Qb = Q  + ((size_t)(b * SEQ + q0)) * DMODEL + h * DK;
  const bf16* Kb = Kp + ((size_t)(b * SEQ)) * DMODEL + h * DK;
  const bf16* Vb = Vt + ((size_t)(b * NHEADS + h)) * DK * SEQ;

  // Q tile fragments for both 32-wide K-steps of d_k=64
  const v16bf a0 = load_a_frag(Qb, DMODEL, 0,  lane);
  const v16bf a1 = load_a_frag(Qb, DMODEL, 32, lane);

  float mrow[8], lrow[8];
  v8f o0 = {}, o1 = {}, o2 = {}, o3 = {};
#pragma unroll
  for (int r = 0; r < 8; ++r) { mrow[r] = -3.0e38f; lrow[r] = 0.0f; }

  const float SCALE = 0.125f;        // 1/sqrt(64)
  const float L2E   = 1.44269504f;

  for (int k0 = 0; k0 < SEQ; k0 += 32) {
    // Prefetch next tile's K rows and V columns while this tile computes.
    if (k0 + 32 < SEQ) {
      __builtin_prefetch(Kb + (size_t)(k0 + 32 + (lane & 31)) * DMODEL, 0, 1);
      __builtin_prefetch(Vb + (size_t)(lane & 31) * SEQ + k0 + 32, 0, 1);
    }

    // ---- QK^T : 16 x 32 scores (two 16x16 accumulators) ----
    v8f s0 = {}, s1 = {};
    s0 = wmma_bf16(a0, load_b_frag(Kb + (size_t)(k0     ) * DMODEL,      DMODEL, lane), s0);
    s0 = wmma_bf16(a1, load_b_frag(Kb + (size_t)(k0     ) * DMODEL + 32, DMODEL, lane), s0);
    s1 = wmma_bf16(a0, load_b_frag(Kb + (size_t)(k0 + 16) * DMODEL,      DMODEL, lane), s1);
    s1 = wmma_bf16(a1, load_b_frag(Kb + (size_t)(k0 + 16) * DMODEL + 32, DMODEL, lane), s1);

    // ---- relative band: G = Q_tile @ Er[rbase .. rbase+47]^T ----
    const int rbase = q0 - k0 + (SEQ - 32);   // in [0, 2000]; rows < ER_PAD
#pragma unroll
    for (int nt = 0; nt < 3; ++nt) {
      const bf16* Eb = Er + (size_t)(rbase + nt * 16) * DK;
      v8f g = {};
      g = wmma_bf16(a0, load_b_frag(Eb,      DK, lane), g);
      g = wmma_bf16(a1, load_b_frag(Eb + 32, DK, lane), g);
#pragma unroll
      for (int r = 0; r < 8; ++r) Gbuf[r + mhi][nt * 16 + nlo] = g[r];
    }
    __syncthreads();   // single-wave WG: cheap barrier

    // ---- combine + online softmax ----
#pragma unroll
    for (int r = 0; r < 8; ++r) {
      const int m = r + mhi;
      float x0 = (s0[r] + Gbuf[m][m - nlo + 31]) * SCALE;  // kk = nlo
      float x1 = (s1[r] + Gbuf[m][m - nlo + 15]) * SCALE;  // kk = 16+nlo
      float t = fmaxf(x0, x1);
      t = fmaxf(t, __shfl_xor(t, 1));
      t = fmaxf(t, __shfl_xor(t, 2));
      t = fmaxf(t, __shfl_xor(t, 4));
      t = fmaxf(t, __shfl_xor(t, 8));
      const float mn = fmaxf(mrow[r], t);
      const float f  = __builtin_exp2f((mrow[r] - mn) * L2E);
      mrow[r] = mn;
      const float e0 = __builtin_exp2f((x0 - mn) * L2E);
      const float e1 = __builtin_exp2f((x1 - mn) * L2E);
      float rs = e0 + e1;
      rs += __shfl_xor(rs, 1);
      rs += __shfl_xor(rs, 2);
      rs += __shfl_xor(rs, 4);
      rs += __shfl_xor(rs, 8);
      lrow[r] = lrow[r] * f + rs;
      o0[r] *= f; o1[r] *= f; o2[r] *= f; o3[r] *= f;
      Pbuf[m][nlo]      = (bf16)e0;
      Pbuf[m][16 + nlo] = (bf16)e1;
    }
    __syncthreads();

    // ---- O += P @ V  (A = P 16x32 from LDS, B = Vt rows contiguous in s) ----
    const v16bf ap = load_a_frag((const bf16*)&Pbuf[0][0], 32, 0, lane);
    o0 = wmma_bf16(ap, load_b_frag(Vb + (size_t) 0 * SEQ + k0, SEQ, lane), o0);
    o1 = wmma_bf16(ap, load_b_frag(Vb + (size_t)16 * SEQ + k0, SEQ, lane), o1);
    o2 = wmma_bf16(ap, load_b_frag(Vb + (size_t)32 * SEQ + k0, SEQ, lane), o2);
    o3 = wmma_bf16(ap, load_b_frag(Vb + (size_t)48 * SEQ + k0, SEQ, lane), o3);
    __syncthreads();
  }

  // ---- normalize and store context (bf16, row-major [b*s, d_model]) ----
  bf16* Cb = Ctx + ((size_t)(b * SEQ + q0)) * DMODEL + h * DK;
#pragma unroll
  for (int r = 0; r < 8; ++r) {
    const float inv = 1.0f / lrow[r];
    const int m = r + mhi;
    Cb[(size_t)m * DMODEL +  0 + nlo] = (bf16)(o0[r] * inv);
    Cb[(size_t)m * DMODEL + 16 + nlo] = (bf16)(o1[r] * inv);
    Cb[(size_t)m * DMODEL + 32 + nlo] = (bf16)(o2[r] * inv);
    Cb[(size_t)m * DMODEL + 48 + nlo] = (bf16)(o3[r] * inv);
  }
}

// ---------------------------------------------------------------------------
// Host-side orchestration
// ---------------------------------------------------------------------------
extern "C" void kernel_launch(void* const* d_in, const int* in_sizes, int n_in,
                              void* d_out, int out_size, void* d_ws, size_t ws_size,
                              hipStream_t stream) {
  (void)in_sizes; (void)n_in; (void)out_size; (void)ws_size;
  const float* q  = (const float*)d_in[0];
  const float* k  = (const float*)d_in[1];
  const float* v  = (const float*)d_in[2];
  const float* Wq = (const float*)d_in[3];
  const float* Wk = (const float*)d_in[4];
  const float* Wv = (const float*)d_in[5];
  const float* Wo = (const float*)d_in[6];
  const float* bq = (const float*)d_in[7];
  const float* bk = (const float*)d_in[8];
  const float* bv = (const float*)d_in[9];
  const float* bo = (const float*)d_in[10];
  const float* Er = (const float*)d_in[11];

  size_t off = 0;
  auto alloc = [&](size_t elems) -> bf16* {
    bf16* p = (bf16*)((char*)d_ws + off);
    off += ((elems * sizeof(bf16)) + 255) & ~(size_t)255;
    return p;
  };
  bf16* Xqb = alloc((size_t)MROWS * DMODEL);
  bf16* Xkb = alloc((size_t)MROWS * DMODEL);
  bf16* Xvb = alloc((size_t)MROWS * DMODEL);
  bf16* Wqb = alloc((size_t)DMODEL * DMODEL);
  bf16* Wkb = alloc((size_t)DMODEL * DMODEL);
  bf16* Wvb = alloc((size_t)DMODEL * DMODEL);
  bf16* Wob = alloc((size_t)DMODEL * DMODEL);
  bf16* Erb = alloc((size_t)ER_PAD * DK);
  bf16* Qb  = alloc((size_t)MROWS * DMODEL);
  bf16* Kb  = alloc((size_t)MROWS * DMODEL);
  bf16* Vtb = alloc((size_t)MROWS * DMODEL);   // [b,h,d,s]
  bf16* Ctx = alloc((size_t)MROWS * DMODEL);

  auto cvt = [&](const float* in, bf16* outp, int nin, int nout) {
    cvt_f32_bf16<<<(nout + 255) / 256, 256, 0, stream>>>(in, outp, nin, nout);
  };
  const int NX = MROWS * DMODEL;
  const int NW = DMODEL * DMODEL;
  cvt(q,  Xqb, NX, NX);
  cvt(k,  Xkb, NX, NX);
  cvt(v,  Xvb, NX, NX);
  cvt(Wq, Wqb, NW, NW);
  cvt(Wk, Wkb, NW, NW);
  cvt(Wv, Wvb, NW, NW);
  cvt(Wo, Wob, NW, NW);
  cvt(Er, Erb, ER_ROWS * DK, ER_PAD * DK);   // pads row 2047 with zeros

  const int GEMM_GRID = (MROWS / 16) * (DMODEL / 64);   // 2048 waves
  gemm16x64_bf16<<<GEMM_GRID, 32, 0, stream>>>(Xqb, Wqb, bq, Qb,  nullptr, 0);
  gemm16x64_bf16<<<GEMM_GRID, 32, 0, stream>>>(Xkb, Wkb, bk, Kb,  nullptr, 0);
  gemm16x64_bf16<<<GEMM_GRID, 32, 0, stream>>>(Xvb, Wvb, bv, Vtb, nullptr, 1);

  const int ATTN_GRID = BATCH * NHEADS * (SEQ / 16);    // 2048 waves
  attn_rel_flash<<<ATTN_GRID, 32, 0, stream>>>(Qb, Kb, Vtb, Erb, Ctx);

  gemm16x64_bf16<<<GEMM_GRID, 32, 0, stream>>>(Ctx, Wob, bo, nullptr, (float*)d_out, 2);
}
